// DeepImageAnalogy_58909771432326
// MI455X (gfx1250) — matrix operand order repaired and checked by reading, not compiled, not run on hardware
//
#include <hip/hip_runtime.h>
#include <hip/hip_bf16.h>

typedef __attribute__((ext_vector_type(16))) _Float16 v16h;
typedef __attribute__((ext_vector_type(8)))  float    v8f;

#define HW   16384   // 128*128
#define CH   256
#define WDIM 128

// ---------------------------------------------------------------------------
// K1: D[p] = sum_c (s[c,p]-t[c,p])^2 via v_wmma_f32_16x16x32_f16.
// One wave handles 16 pixels; A = diff tile [16 x 32] (f16), B = A^T.
// Diagonal of A*A^T (f32 accum over 8 K-chunks) gives the 16 D values.
// ---------------------------------------------------------------------------
__global__ __launch_bounds__(128) void k_diffD(const float* __restrict__ s,
                                               const float* __restrict__ t,
                                               float* __restrict__ D) {
    const int lane = threadIdx.x & 31;
    const int wave = threadIdx.x >> 5;
    const int tile = blockIdx.x * 4 + wave;        // 1024 tiles of 16 pixels
    const int p0   = tile * 16;
    const int m    = lane & 15;
    const int p    = p0 + m;
    const int hi   = lane >> 4;                    // 0: lanes 0-15, 1: lanes 16-31
    const int khalf = hi * 8;

    v8f c = {};
    for (int c0 = 0; c0 < CH; c0 += 32) {
        float dv[32];
#pragma unroll
        for (int j = 0; j < 32; ++j) {
            const int k = c0 + j;
            dv[j] = s[k * HW + p] - t[k * HW + p];
        }
        // A layout (16-bit A 16x32, ISA table): lane-lo halves = K{0..7,16..23},
        // lane-hi halves = K{8..15,24..31}; M = lane&15.
        v16h a, b;
#pragma unroll
        for (int j = 0; j < 8; ++j) {
            a[j]     = (_Float16)dv[khalf + j];
            a[8 + j] = (_Float16)dv[16 + khalf + j];
        }
        // B layout (same K order as A): lane-lo halves = K{0..7,16..23}, lane-hi = K{8..15,24..31};
        // N = lane&15.  B = A^T -> B[K][N] = diff[K][pixel N] = dv[K] at this lane.
#pragma unroll
        for (int j = 0; j < 16; ++j) {
            b[j] = (_Float16)dv[j < 8 ? khalf + j : 16 + khalf + (j - 8)];
        }
        c = __builtin_amdgcn_wmma_f32_16x16x32_f16(
            /*neg_a=*/false, a, /*neg_b=*/false, b,
            /*c_mod=*/(short)0, c, /*reuse_a=*/false, /*reuse_b=*/false);
    }
    // Diagonal: element g of lane g holds D[g]; element g of lane g+24 holds D[g+8].
    float dval = 0.f;
#pragma unroll
    for (int g = 0; g < 8; ++g) {
        if (lane == g)      dval = c[g];
        if (lane == g + 24) dval = c[g];
    }
    if (lane < 8)       D[p0 + lane]       = dval;
    else if (lane >= 24) D[p0 + lane - 16] = dval;
}

// ---------------------------------------------------------------------------
// K2: S[y,x] = sum_{dy,dx in [-1,1]} D[clip(y+dy)][clip(x+dx)]
// ---------------------------------------------------------------------------
__global__ __launch_bounds__(256) void k_box(const float* __restrict__ D,
                                             float* __restrict__ S) {
    const int p = blockIdx.x * 256 + threadIdx.x;
    const int y = p >> 7, x = p & 127;
    float acc = 0.f;
#pragma unroll
    for (int dy = -1; dy <= 1; ++dy) {
        int yy = y + dy; yy = yy < 0 ? 0 : (yy > 127 ? 127 : yy);
#pragma unroll
        for (int dx = -1; dx <= 1; ++dx) {
            int xx = x + dx; xx = xx < 0 ? 0 : (xx > 127 ? 127 : xx);
            acc += D[(yy << 7) | xx];
        }
    }
    S[p] = acc;
}

// ---------------------------------------------------------------------------
// K3: full PatchMatch.  Rows are independent (roll is circular along x only),
// so each block owns 2 rows (256 threads, 1 px/thread), nnf in registers,
// neighbor exchange via LDS, S table (64KB) in dynamic LDS.
// gridDim.y: 0 = forward, 1 = backward.
// ---------------------------------------------------------------------------
__device__ __forceinline__ float lookupS(const float* sS, float fy, float fx) {
    int iy = (int)__builtin_rintf(fy);   // round-half-even, matches jnp.round
    int ix = (int)__builtin_rintf(fx);
    iy = iy < 0 ? 0 : (iy > 127 ? 127 : iy);
    ix = ix < 0 ? 0 : (ix > 127 ? 127 : ix);
    return sS[(iy << 7) | ix];
}

__global__ __launch_bounds__(256) void k_patchmatch(
    const float* __restrict__ S,
    const float* __restrict__ init_f, const float* __restrict__ init_b,
    const float* __restrict__ rand_f, const float* __restrict__ rand_b,
    int* __restrict__ iyF, int* __restrict__ ixF,
    float* __restrict__ dF, float* __restrict__ dB) {
    extern __shared__ float smem[];
    float* sS = smem;                 // 16384 floats
    float* ly = smem + HW;            // 256 floats
    float* lx = smem + HW + 256;      // 256 floats

    const int tid = threadIdx.x;
    for (int i = tid; i < HW; i += 256) sS[i] = S[i];

    const bool fwd = (blockIdx.y == 0);
    const float* init = fwd ? init_f : init_b;
    const float* rnd  = fwd ? rand_f : rand_b;

    const int ty = tid >> 7;                 // row within block (0/1)
    const int x  = tid & 127;
    const int r  = blockIdx.x * 2 + ty;      // global row
    const int p  = (r << 7) | x;

    float ny = init[p]      * 127.0f;        // scale = Ht-1 = 127
    float nx = init[HW + p] * 127.0f;
    __syncthreads();

    const float scales[4] = {1.0f, 0.5f, 0.25f, 0.125f};

    for (int it = 0; it < 5; ++it) {
        // propagation, direction = +1 (candidate from x-1 mod W, OLD values)
        ly[tid] = ny; lx[tid] = nx;
        __syncthreads();
        {
            const int n = (ty << 7) | ((x + 127) & 127);
            const float cy = ly[n], cx = lx[n];
            if (lookupS(sS, cy, cx) < lookupS(sS, ny, nx)) { ny = cy; nx = cx; }
        }
        __syncthreads();
        // propagation, direction = -1 (candidate from x+1 mod W)
        ly[tid] = ny; lx[tid] = nx;
        __syncthreads();
        {
            const int n = (ty << 7) | ((x + 1) & 127);
            const float cy = ly[n], cx = lx[n];
            if (lookupS(sS, cy, cx) < lookupS(sS, ny, nx)) { ny = cy; nx = cx; }
        }
        __syncthreads();
        // random search (purely per-pixel)
        float dcur = lookupS(sS, ny, nx);
#pragma unroll
        for (int i = 0; i < 4; ++i) {
            const float* rp = rnd + (size_t)((it * 4 + i) * 2) * HW;
            float ry = ny + rp[p]      * scales[i];
            float rx = nx + rp[HW + p] * scales[i];
            ry = fminf(fmaxf(ry, 0.0f), 127.0f);
            rx = fminf(fmaxf(rx, 0.0f), 127.0f);
            const float d = lookupS(sS, ry, rx);
            if (d < dcur) { ny = ry; nx = rx; dcur = d; }
        }
    }

    const float dfin = lookupS(sS, ny, nx);
    if (fwd) {
        int iy = (int)__builtin_rintf(ny); iy = iy < 0 ? 0 : (iy > 127 ? 127 : iy);
        int ix = (int)__builtin_rintf(nx); ix = ix < 0 ? 0 : (ix > 127 ? 127 : ix);
        iyF[p] = iy; ixF[p] = ix; dF[p] = dfin;
    } else {
        dB[p] = dfin;
    }
}

// ---------------------------------------------------------------------------
// K4: similarity = -0.5 * (S_at(nnf_fwd) + S_at(nnf_bwd))
// ---------------------------------------------------------------------------
__global__ __launch_bounds__(256) void k_sim(const float* __restrict__ dF,
                                             const float* __restrict__ dB,
                                             float* __restrict__ sim) {
    const int p = blockIdx.x * 256 + threadIdx.x;
    sim[p] = -0.5f * (dF[p] + dB[p]);
}

// ---------------------------------------------------------------------------
// K5: reconstruct.  For output pixel (oy,ox): sum over the <=9 neighbors
// (sy,sx)=(oy-u,ox-v) of gather[clip(iy[n]+u)][clip(ix[n]+v)], / cnt, blend.
// gridDim.y: 0 = rec_s (gather target, coords as-is),
//            1 = rec_t (gather source, coords swapped: nnf[:, ::-1]).
// Wave = 32 consecutive pixels (coalesced base/out); 8 waves = 8 channel groups.
// ---------------------------------------------------------------------------
__global__ __launch_bounds__(256) void k_recon(const float* __restrict__ src_f,
                                               const float* __restrict__ tgt_f,
                                               const int* __restrict__ iyF,
                                               const int* __restrict__ ixF,
                                               float* __restrict__ out) {
    const bool is_s = (blockIdx.y == 0);
    const float* baseF = is_s ? src_f : tgt_f;
    const float* gat   = is_s ? tgt_f : src_f;
    float* o = out + (is_s ? 0 : (size_t)CH * HW);

    const int lane = threadIdx.x & 31;
    const int cg   = threadIdx.x >> 5;            // channel group 0..7
    const int p    = blockIdx.x * 32 + lane;
    const int oy   = p >> 7, ox = p & 127;

    int   addr[9];
    float w[9];
    int   cnt = 0;
#pragma unroll
    for (int u = -1; u <= 1; ++u) {
#pragma unroll
        for (int v = -1; v <= 1; ++v) {
            const int sy = oy - u, sx = ox - v;
            const bool ok = (sy >= 0) & (sy < 128) & (sx >= 0) & (sx < 128);
            const int np = ok ? ((sy << 7) | sx) : 0;
            int gy = iyF[np], gx = ixF[np];
            if (!is_s) { const int tmp = gy; gy = gx; gx = tmp; }  // nnf[:, ::-1]
            int tyc = gy + u; tyc = tyc < 0 ? 0 : (tyc > 127 ? 127 : tyc);
            int txc = gx + v; txc = txc < 0 ? 0 : (txc > 127 ? 127 : txc);
            const int j = (u + 1) * 3 + (v + 1);
            addr[j] = (tyc << 7) | txc;
            w[j]    = ok ? 1.0f : 0.0f;
            cnt    += ok ? 1 : 0;
        }
    }
    const float inv = 0.6f / ((float)cnt + 1e-8f);

    for (int cc = 0; cc < 32; ++cc) {
        const int c = cg * 32 + cc;
        const float* g = gat + (size_t)c * HW;
        float acc = 0.f;
#pragma unroll
        for (int j = 0; j < 9; ++j) acc = fmaf(w[j], g[addr[j]], acc);
        o[(size_t)c * HW + p] = 0.4f * baseF[(size_t)c * HW + p] + inv * acc;
    }
}

// ---------------------------------------------------------------------------
extern "C" void kernel_launch(void* const* d_in, const int* in_sizes, int n_in,
                              void* d_out, int out_size, void* d_ws, size_t ws_size,
                              hipStream_t stream) {
    (void)in_sizes; (void)n_in; (void)out_size; (void)ws_size;
    const float* src    = (const float*)d_in[0];   // [1,256,128,128]
    const float* tgt    = (const float*)d_in[1];   // [1,256,128,128]
    const float* init_f = (const float*)d_in[2];   // [1,2,128,128]
    const float* init_b = (const float*)d_in[3];   // [1,2,128,128]
    const float* rand_f = (const float*)d_in[4];   // [5,4,1,2,128,128]
    const float* rand_b = (const float*)d_in[5];   // [5,4,1,2,128,128]

    float* out   = (float*)d_out;
    float* rec   = out;                            // rec_s, then rec_t
    float* sim   = out + 2 * (size_t)CH * HW;      // similarity

    // workspace carve (floats/ints, 16384 elems each)
    float* wsf = (float*)d_ws;
    float* D   = wsf;
    float* S   = wsf + HW;
    int*   iyF = (int*)(wsf + 2 * HW);
    int*   ixF = (int*)(wsf + 3 * HW);
    float* dF  = wsf + 4 * HW;
    float* dB  = wsf + 5 * HW;

    // K1: per-pixel squared feature distance via WMMA (1024 waves, 4 waves/block)
    k_diffD<<<dim3(256), dim3(128), 0, stream>>>(src, tgt, D);
    // K2: 3x3 clamped box sum
    k_box<<<dim3(64), dim3(256), 0, stream>>>(D, S);
    // K3: forward + backward PatchMatch (2 rows/block, dynamic LDS: S + row bufs)
    const size_t lds = (HW + 512) * sizeof(float);  // 67584 B
    k_patchmatch<<<dim3(64, 2), dim3(256), lds, stream>>>(
        S, init_f, init_b, rand_f, rand_b, iyF, ixF, dF, dB);
    // K4: similarity
    k_sim<<<dim3(64), dim3(256), 0, stream>>>(dF, dB, sim);
    // K5: rec_s and rec_t
    k_recon<<<dim3(512, 2), dim3(256), 0, stream>>>(src, tgt, iyF, ixF, rec);
}
